// MyNetworkMapper_14869176779412
// MI455X (gfx1250) — matrix operand order repaired and checked by reading, not compiled, not run on hardware
//
#include <hip/hip_runtime.h>

// ===========================================================================
// GNN message passing on gfx1250 (MI455X), bf16 WMMA everywhere.
//
// H=64 hidden dim maps onto V_WMMA_F32_16X16X32_BF16: each wave32 computes a
// 16-row x 64-col output tile (4 x v8f accumulators), K-loop in steps of 32.
// Weights are pre-converted once per launch into "fragment-linear" bf16 so a
// wave loads each 32x16 B tile with two 16B loads per lane.
// Edge gathers go straight to LDS via GLOBAL_LOAD_ASYNC_TO_LDS_B128.
// fp32->bf16 uses v_cvt_pk_bf16_f32 via inline asm (no clang builtin on this
// toolchain).
// ===========================================================================

typedef __attribute__((ext_vector_type(16))) __bf16 v16bf;
typedef __attribute__((ext_vector_type(8)))  float  v8f;

// Pointer types for the async-to-LDS builtin (deduced from hipcc diagnostic:
// param 0 is AS1 pointer to int __attribute__((vector_size(16)))).
typedef int v4i_t __attribute__((vector_size(16)));
typedef __attribute__((address_space(1))) v4i_t* gv4i_p;  // global
typedef __attribute__((address_space(3))) v4i_t* lv4i_p;  // LDS

union Frag32B { v16bf v; uint4 q[2]; unsigned short h[16]; };

#if defined(__has_builtin)
#  if __has_builtin(__builtin_amdgcn_global_load_async_to_lds_b128)
#    define HAVE_ASYNC_LDS 1
#  endif
#endif

// Pack two fp32 into one dword of bf16x2 with a single VALU op.
__device__ __forceinline__ unsigned pack2bf(float a, float b) {
  unsigned r;
  asm("v_cvt_pk_bf16_f32 %0, %1, %2" : "=v"(r) : "v"(a), "v"(b));
  return r;
}

__device__ __forceinline__ unsigned short f2bf(float f) {
  return (unsigned short)pack2bf(f, f);
}

__device__ __forceinline__ float reluf(float v) { return __builtin_fmaxf(v, 0.f); }

__device__ __forceinline__ v8f wmma_bf16(v16bf a, v16bf b, v8f c) {
  // (neg_a, A, neg_b, B, c_mod, C, reuse_a, reuse_b)
  return __builtin_amdgcn_wmma_f32_16x16x32_bf16(false, a, false, b,
                                                 (short)0, c, false, false);
}

// B fragment: fragment-linear layout, tile = kt*Ntiles + nt, 512 halves/tile,
// lane-major (lane*16 + h). Matches 32x16 KxN VGPR striping:
//   lane: N = lane&15, K = kt*32 + (lane>>4)*8 + (h&7) + (h>>3)*16
__device__ __forceinline__ v16bf load_frag_b(const uint4* wfrag, int tile, int lane) {
  Frag32B f;
  const uint4* p = wfrag + (size_t)tile * 64 + lane * 2;
  f.q[0] = p[0];
  f.q[1] = p[1];
  return f.v;
}

// A fragment from an LDS tile of bf16 halves, row stride strideH (mult of 8).
// ISA 16-bit A 16x32 layout: lane (m = lane&15, hi = lane>>4) holds
// K = ktBase + hi*8 + {0..7} in halves 0..7 and +16 in halves 8..15.
__device__ __forceinline__ v16bf load_frag_a(const unsigned short* base, int strideH,
                                             int ktBase, int lane) {
  int m  = lane & 15;
  int hi = lane >> 4;
  const unsigned short* p = base + m * strideH + ktBase + hi * 8;
  Frag32B f;
  f.q[0] = *reinterpret_cast<const uint4*>(p);
  f.q[1] = *reinterpret_cast<const uint4*>(p + 16);
  return f.v;
}

struct MlpW {
  const uint4* w1; const float* b1;
  const uint4* w2; const float* b2;
  const float* g;  const float* be;
};

#define H1S 72   // h1 LDS tile stride (halves)
#define LNS 68   // layernorm LDS tile stride (floats) -> conflict-free rows

// Linear(K1->64) + ReLU + Linear(64->64) + ReLU + LayerNorm.
// Result: normalized fp32 rows in lnB (16 x LNS). Ends with __syncthreads().
template<int NK1>
__device__ __forceinline__ void mlp_ln_to_lds(const unsigned short* inT, int inS,
                                              MlpW W, unsigned short* h1T,
                                              float* lnB, int lane) {
  v8f acc[4] = {};
  #pragma unroll
  for (int kt = 0; kt < NK1; ++kt) {
    v16bf a = load_frag_a(inT, inS, kt * 32, lane);
    #pragma unroll
    for (int nt = 0; nt < 4; ++nt)
      acc[nt] = wmma_bf16(a, load_frag_b(W.w1, kt * 4 + nt, lane), acc[nt]);
  }
  int hi = lane >> 4, colq = lane & 15;
  #pragma unroll
  for (int nt = 0; nt < 4; ++nt) {
    float bb = W.b1[nt * 16 + colq];
    #pragma unroll
    for (int r = 0; r < 8; r += 2) {
      // convert two rows with one v_cvt_pk_bf16_f32, store halves separately
      unsigned pk = pack2bf(reluf(acc[nt][r] + bb), reluf(acc[nt][r + 1] + bb));
      h1T[(r + 0 + hi * 8) * H1S + nt * 16 + colq] = (unsigned short)pk;
      h1T[(r + 1 + hi * 8) * H1S + nt * 16 + colq] = (unsigned short)(pk >> 16);
    }
  }
  __syncthreads();
  v8f acc2[4] = {};
  #pragma unroll
  for (int kt = 0; kt < 2; ++kt) {
    v16bf a = load_frag_a(h1T, H1S, kt * 32, lane);
    #pragma unroll
    for (int nt = 0; nt < 4; ++nt)
      acc2[nt] = wmma_bf16(a, load_frag_b(W.w2, kt * 4 + nt, lane), acc2[nt]);
  }
  #pragma unroll
  for (int nt = 0; nt < 4; ++nt) {
    float bb = W.b2[nt * 16 + colq];
    #pragma unroll
    for (int r = 0; r < 8; ++r)
      lnB[(r + hi * 8) * LNS + nt * 16 + colq] = reluf(acc2[nt][r] + bb);
  }
  __syncthreads();
  if (lane < 16) {                 // one lane per row; padded stride -> no conflicts
    float* rowp = lnB + lane * LNS;
    float s = 0.f, s2 = 0.f;
    #pragma unroll
    for (int j = 0; j < 64; ++j) { float v = rowp[j]; s += v; s2 += v * v; }
    float m   = s  * (1.f / 64.f);
    float var = s2 * (1.f / 64.f) - m * m;
    float rstd = rsqrtf(var + 1e-5f);
    #pragma unroll
    for (int j = 0; j < 64; ++j)
      rowp[j] = (rowp[j] - m) * rstd * W.g[j] + W.be[j];
  }
  __syncthreads();
}

// Store one half-row (32 floats) as bf16 with 4 x b128 stores.
__device__ __forceinline__ void store_row_bf16(unsigned short* dst, const float* s) {
  #pragma unroll
  for (int q = 0; q < 4; ++q) {
    uint4 d;
    d.x = pack2bf(s[q * 8 + 0], s[q * 8 + 1]);
    d.y = pack2bf(s[q * 8 + 2], s[q * 8 + 3]);
    d.z = pack2bf(s[q * 8 + 4], s[q * 8 + 5]);
    d.w = pack2bf(s[q * 8 + 6], s[q * 8 + 7]);
    *reinterpret_cast<uint4*>(dst + q * 8) = d;
  }
}

// ---------------------------------------------------------------------------
// Weight prep: fp32 [Kin][Nin] row-major -> bf16 fragment-linear, zero padded
// to Ktiles*32 x Ntiles*16. swapK swaps K-blocks [0,64) <-> [64,128) so the
// edge-update pass (cat order [x_row, x_col, e]) can reuse the gathered
// [x_col, x_row, e] input tile of the message pass.
// ---------------------------------------------------------------------------
__global__ void prep_weights(const float* __restrict__ W, unsigned short* __restrict__ out,
                             int Kin, int Nin, int Ktiles, int Ntiles, int swapK) {
  int tid = blockIdx.x * 256 + threadIdx.x;
  int total = Ktiles * Ntiles * 512;
  if (tid >= total) return;
  int tile = tid >> 9, w = tid & 511;
  int lane = w >> 4, h = w & 15;
  int kt = tile / Ntiles, nt = tile % Ntiles;
  int K = kt * 32 + ((lane >> 4) << 3) + (h & 7) + ((h >> 3) << 4);
  int N = nt * 16 + (lane & 15);
  int sk = K;
  if (swapK) sk = (K < 64) ? K + 64 : (K < 128 ? K - 64 : K);
  float v = (sk < Kin && N < Nin) ? W[(size_t)sk * Nin + N] : 0.f;
  out[tid] = f2bf(v);
}

__global__ void zero_f32(float* p, int n) {
  int i = blockIdx.x * 256 + threadIdx.x;
  if (i < n) p[i] = 0.f;
}

// ---------------------------------------------------------------------------
// Encoders: fp32 [rows][Kin] (Kin<=16, K zero-padded to 32) -> mlp_ln -> bf16
// ---------------------------------------------------------------------------
__global__ __launch_bounds__(128) void encode_kernel(
    const float* __restrict__ in, int nRows, int Kin, MlpW W,
    unsigned short* __restrict__ out) {
  __shared__ unsigned short sIn[4][16 * 40];
  __shared__ unsigned short sH1[4][16 * H1S];
  __shared__ float          sLn[4][16 * LNS];
  int wv = threadIdx.x >> 5, lane = threadIdx.x & 31;
  int rb = (blockIdx.x * 4 + wv) * 16;
  if (lane < 16) {
    int r = rb + lane;
    bool ok = r < nRows;
    const float* src = in + (size_t)(ok ? r : 0) * Kin;
    for (int j = 0; j < 32; j += 2) {
      float a = (ok && j + 0 < Kin) ? src[j + 0] : 0.f;
      float b = (ok && j + 1 < Kin) ? src[j + 1] : 0.f;
      *reinterpret_cast<unsigned*>(&sIn[wv][lane * 40 + j]) = pack2bf(a, b);
    }
  }
  __syncthreads();
  mlp_ln_to_lds<1>(sIn[wv], 40, W, sH1[wv], sLn[wv], lane);
  int row = lane & 15, hf = lane >> 4;
  int r = rb + row;
  if (r < nRows)
    store_row_bf16(out + (size_t)r * 64 + hf * 32, sLn[wv] + row * LNS + hf * 32);
}

// ---------------------------------------------------------------------------
// Message + edge-update kernel. One wave per 16 edges. Gathers
// [xh[col] | xh[row] | eh] (192 bf16) into LDS once, then:
//   pass 1: edge_net (normal W1)    -> layernorm -> atomic f32 add to aggr[col]
//   pass 2: edge_net (K-swapped W1) -> layernorm -> eh_new
// ---------------------------------------------------------------------------
#define MSGS 200  // input tile stride in halves (192 + 8 pad)
__global__ __launch_bounds__(128) void msg_kernel(
    const int* __restrict__ ei, int E,
    const unsigned short* __restrict__ xh, const unsigned short* __restrict__ eh,
    MlpW Wm, const uint4* __restrict__ w1_swapped,
    float* __restrict__ aggr, unsigned short* __restrict__ ehNew) {
  __shared__ unsigned short sIn[4][16 * MSGS];
  __shared__ unsigned short sH1[4][16 * H1S];
  __shared__ float          sLn[4][16 * LNS];
  __shared__ int            sRowI[4][16], sColI[4][16];
  int wv = threadIdx.x >> 5, lane = threadIdx.x & 31;
  int eb = (blockIdx.x * 4 + wv) * 16;

  if (lane < 16) {
    int e = eb + lane;
    bool ok = e < E;
    sRowI[wv][lane] = ok ? ei[e]     : 0;   // source  (x_j)
    sColI[wv][lane] = ok ? ei[E + e] : 0;   // target  (x_i / aggregation index)
  }
  __syncthreads();

  // Gather: 16 edges x 3 segments x 8 x 16B chunks = 384 b128 transfers,
  // 12 per lane, straight into LDS via async DMA when available.
  for (int t = 0; t < 12; ++t) {
    int u = t * 32 + lane;
    int el = u / 24, c = u % 24;
    int seg = c >> 3, off = c & 7;
    const unsigned short* src;
    if (seg == 0)      src = xh + (size_t)sColI[wv][el] * 64 + off * 8;
    else if (seg == 1) src = xh + (size_t)sRowI[wv][el] * 64 + off * 8;
    else {
      int e = eb + el; if (e >= E) e = 0;
      src = eh + (size_t)e * 64 + off * 8;
    }
    unsigned short* dst = &sIn[wv][el * MSGS + seg * 64 + off * 8];
#ifdef HAVE_ASYNC_LDS
    __builtin_amdgcn_global_load_async_to_lds_b128((gv4i_p)src, (lv4i_p)dst, 0, 0);
#else
    *reinterpret_cast<uint4*>(dst) = *reinterpret_cast<const uint4*>(src);
#endif
  }
#ifdef HAVE_ASYNC_LDS
#  if __has_builtin(__builtin_amdgcn_s_wait_asynccnt)
  __builtin_amdgcn_s_wait_asynccnt(0);
#  else
  asm volatile("s_wait_asynccnt 0x0" ::: "memory");
#  endif
#endif
  __syncthreads();

  // ---- pass 1: messages -> segment-sum via f32 atomics (wave-wide) ----
  mlp_ln_to_lds<6>(sIn[wv], MSGS, Wm, sH1[wv], sLn[wv], lane);
  {
    int row = lane & 15, hf = lane >> 4;
    int e = eb + row;
    if (e < E) {
      int cn = sColI[wv][row];
      float* dst = aggr + (size_t)cn * 64 + hf * 32;
      const float* s = sLn[wv] + row * LNS + hf * 32;
      #pragma unroll
      for (int j = 0; j < 32; ++j) atomicAdd(dst + j, s[j]);
    }
  }
  __syncthreads();

  // ---- pass 2: edge update, same input tile, K-swapped L1 weights ----
  MlpW Ws = Wm; Ws.w1 = w1_swapped;
  mlp_ln_to_lds<6>(sIn[wv], MSGS, Ws, sH1[wv], sLn[wv], lane);
  {
    int row = lane & 15, hf = lane >> 4;
    int e = eb + row;
    if (e < E)
      store_row_bf16(ehNew + (size_t)e * 64 + hf * 32, sLn[wv] + row * LNS + hf * 32);
  }
}

// ---------------------------------------------------------------------------
// Node update: cat([aggr(fp32->bf16), xh]) -> node_net mlp_ln -> xh_new
// ---------------------------------------------------------------------------
__global__ __launch_bounds__(128) void node_update_kernel(
    int Nn, const float* __restrict__ aggr, const unsigned short* __restrict__ xh,
    MlpW W, unsigned short* __restrict__ xhNew) {
  __shared__ unsigned short sIn[4][16 * 136];
  __shared__ unsigned short sH1[4][16 * H1S];
  __shared__ float          sLn[4][16 * LNS];
  int wv = threadIdx.x >> 5, lane = threadIdx.x & 31;
  int nb = (blockIdx.x * 4 + wv) * 16;
  if (lane < 16) {
    int n = nb + lane;
    int ns = (n < Nn) ? n : 0;
    const float* a = aggr + (size_t)ns * 64;
    for (int j = 0; j < 64; j += 2) {
      unsigned p = pack2bf(a[j], a[j + 1]);
      *reinterpret_cast<unsigned*>(&sIn[wv][lane * 136 + j]) = p;
    }
    const uint4* xr = reinterpret_cast<const uint4*>(xh + (size_t)ns * 64);
    #pragma unroll
    for (int q = 0; q < 8; ++q)
      *reinterpret_cast<uint4*>(&sIn[wv][lane * 136 + 64 + q * 8]) = xr[q];
  }
  __syncthreads();
  mlp_ln_to_lds<4>(sIn[wv], 136, W, sH1[wv], sLn[wv], lane);
  int row = lane & 15, hf = lane >> 4;
  int n = nb + row;
  if (n < Nn)
    store_row_bf16(xhNew + (size_t)n * 64 + hf * 32, sLn[wv] + row * LNS + hf * 32);
}

// ---------------------------------------------------------------------------
// Final head: relu(xh @ W1 + b1) @ W2[64x3] + b2 -> fp32 out (N x 3).
// W2 N-padded to 16; only cols 0..2 stored.
// ---------------------------------------------------------------------------
__global__ __launch_bounds__(128) void final_kernel(
    int Nn, const unsigned short* __restrict__ xh,
    const uint4* __restrict__ w1, const float* __restrict__ b1,
    const uint4* __restrict__ w2, const float* __restrict__ b2,
    float* __restrict__ out) {
  __shared__ unsigned short sH1[4][16 * H1S];
  int wv = threadIdx.x >> 5, lane = threadIdx.x & 31;
  int nb = (blockIdx.x * 4 + wv) * 16;
  int hi = lane >> 4, colq = lane & 15;

  v8f acc[4] = {};
  #pragma unroll
  for (int kt = 0; kt < 2; ++kt) {
    int m = nb + colq; if (m >= Nn) m = 0;     // per-lane row clamp
    const unsigned short* p = xh + (size_t)m * 64 + kt * 32 + hi * 8;
    Frag32B f;
    f.q[0] = *reinterpret_cast<const uint4*>(p);
    f.q[1] = *reinterpret_cast<const uint4*>(p + 16);
    #pragma unroll
    for (int nt = 0; nt < 4; ++nt)
      acc[nt] = wmma_bf16(f.v, load_frag_b(w1, kt * 4 + nt, lane), acc[nt]);
  }
  #pragma unroll
  for (int nt = 0; nt < 4; ++nt) {
    float bb = b1[nt * 16 + colq];
    #pragma unroll
    for (int r = 0; r < 8; r += 2) {
      unsigned pk = pack2bf(reluf(acc[nt][r] + bb), reluf(acc[nt][r + 1] + bb));
      sH1[wv][(r + 0 + hi * 8) * H1S + nt * 16 + colq] = (unsigned short)pk;
      sH1[wv][(r + 1 + hi * 8) * H1S + nt * 16 + colq] = (unsigned short)(pk >> 16);
    }
  }
  __syncthreads();
  v8f o = {};
  #pragma unroll
  for (int kt = 0; kt < 2; ++kt) {
    v16bf a = load_frag_a(sH1[wv], H1S, kt * 32, lane);
    o = wmma_bf16(a, load_frag_b(w2, kt, lane), o);  // Ntiles == 1
  }
  if (colq < 3) {
    float bb = b2[colq];
    #pragma unroll
    for (int r = 0; r < 8; ++r) {
      int row = nb + r + hi * 8;
      if (row < Nn) out[(size_t)row * 3 + colq] = o[r] + bb;
    }
  }
}

// ===========================================================================
// Host side
// ===========================================================================
extern "C" void kernel_launch(void* const* d_in, const int* in_sizes, int n_in,
                              void* d_out, int out_size, void* d_ws, size_t ws_size,
                              hipStream_t stream) {
  // Flattened input order (depth-first, dict insertion order of setup_inputs):
  //  0: x [N,16] f32   1: edge_index [2,E] i32   2: edge_attr [E,8] f32
  //  then params: node0{l1.w,l1.b,l2.w,l2.b,g,be}, edge0{...}, edge_net{...},
  //  node_net{...}, final{l1.w,l1.b,l2.w,l2.b}
  const float* x  = (const float*)d_in[0];
  const int*   ei = (const int*)d_in[1];
  const float* ea = (const float*)d_in[2];
  int idx = 3;
  auto nf = [&]() { return (const float*)d_in[idx++]; };
  const float *n0w1 = nf(), *n0b1 = nf(), *n0w2 = nf(), *n0b2 = nf(), *n0g = nf(), *n0be = nf();
  const float *e0w1 = nf(), *e0b1 = nf(), *e0w2 = nf(), *e0b2 = nf(), *e0g = nf(), *e0be = nf();
  const float *enw1 = nf(), *enb1 = nf(), *enw2 = nf(), *enb2 = nf(), *eng = nf(), *enbe = nf();
  const float *nnw1 = nf(), *nnb1 = nf(), *nnw2 = nf(), *nnb2 = nf(), *nng = nf(), *nnbe = nf();
  const float *fw1  = nf(), *fb1  = nf(), *fw2  = nf(), *fb2  = nf();

  const int Nn = in_sizes[0] / 16;   // 50000
  const int E  = in_sizes[1] / 2;    // 800000

  // Workspace carve-out
  char* ws = (char*)d_ws;
  size_t off = 0;
  auto alloc = [&](size_t bytes) -> char* {
    char* p = ws + off;
    off = (off + bytes + 255) & ~(size_t)255;
    return p;
  };
  unsigned short* wN0L1  = (unsigned short*)alloc(1 * 4 * 512 * 2);
  unsigned short* wN0L2  = (unsigned short*)alloc(2 * 4 * 512 * 2);
  unsigned short* wE0L1  = (unsigned short*)alloc(1 * 4 * 512 * 2);
  unsigned short* wE0L2  = (unsigned short*)alloc(2 * 4 * 512 * 2);
  unsigned short* wENL1  = (unsigned short*)alloc(6 * 4 * 512 * 2);
  unsigned short* wENL1s = (unsigned short*)alloc(6 * 4 * 512 * 2);
  unsigned short* wENL2  = (unsigned short*)alloc(2 * 4 * 512 * 2);
  unsigned short* wNNL1  = (unsigned short*)alloc(4 * 4 * 512 * 2);
  unsigned short* wNNL2  = (unsigned short*)alloc(2 * 4 * 512 * 2);
  unsigned short* wFL1   = (unsigned short*)alloc(2 * 4 * 512 * 2);
  unsigned short* wFL2   = (unsigned short*)alloc(2 * 1 * 512 * 2);
  unsigned short* xhA  = (unsigned short*)alloc((size_t)Nn * 64 * 2);
  unsigned short* xhB  = (unsigned short*)alloc((size_t)Nn * 64 * 2);
  unsigned short* ehA  = (unsigned short*)alloc((size_t)E * 64 * 2);
  unsigned short* ehB  = (unsigned short*)alloc((size_t)E * 64 * 2);
  float*          aggr = (float*)alloc((size_t)Nn * 64 * 4);
  (void)ws_size; (void)n_in; (void)out_size;

  auto prep = [&](const float* W, unsigned short* dst, int Kin, int Nin,
                  int Kt, int Nt, int swap) {
    int total = Kt * Nt * 512;
    prep_weights<<<(total + 255) / 256, 256, 0, stream>>>(W, dst, Kin, Nin, Kt, Nt, swap);
  };
  prep(n0w1, wN0L1,  16, 64, 1, 4, 0);
  prep(n0w2, wN0L2,  64, 64, 2, 4, 0);
  prep(e0w1, wE0L1,   8, 64, 1, 4, 0);
  prep(e0w2, wE0L2,  64, 64, 2, 4, 0);
  prep(enw1, wENL1, 192, 64, 6, 4, 0);
  prep(enw1, wENL1s,192, 64, 6, 4, 1);   // K blocks [0,64)<->[64,128) swapped
  prep(enw2, wENL2,  64, 64, 2, 4, 0);
  prep(nnw1, wNNL1, 128, 64, 4, 4, 0);
  prep(nnw2, wNNL2,  64, 64, 2, 4, 0);
  prep(fw1,  wFL1,   64, 64, 2, 4, 0);
  prep(fw2,  wFL2,   64,  3, 2, 1, 0);

  MlpW Wn0 { (const uint4*)wN0L1, n0b1, (const uint4*)wN0L2, n0b2, n0g, n0be };
  MlpW We0 { (const uint4*)wE0L1, e0b1, (const uint4*)wE0L2, e0b2, e0g, e0be };
  MlpW Wen { (const uint4*)wENL1, enb1, (const uint4*)wENL2, enb2, eng, enbe };
  MlpW Wnn { (const uint4*)wNNL1, nnb1, (const uint4*)wNNL2, nnb2, nng, nnbe };

  const int tilesN = (Nn + 15) / 16, wgsN = (tilesN + 3) / 4;
  const int tilesE = (E  + 15) / 16, wgsE = (tilesE + 3) / 4;

  encode_kernel<<<wgsN, 128, 0, stream>>>(x,  Nn, 16, Wn0, xhA);
  encode_kernel<<<wgsE, 128, 0, stream>>>(ea, E,   8, We0, ehA);

  unsigned short *xhCur = xhA, *xhNxt = xhB, *ehCur = ehA, *ehNxt = ehB;
  for (int r = 0; r < 3; ++r) {
    zero_f32<<<(Nn * 64 + 255) / 256, 256, 0, stream>>>(aggr, Nn * 64);
    msg_kernel<<<wgsE, 128, 0, stream>>>(ei, E, xhCur, ehCur, Wen,
                                         (const uint4*)wENL1s, aggr, ehNxt);
    node_update_kernel<<<wgsN, 128, 0, stream>>>(Nn, aggr, xhCur, Wnn, xhNxt);
    unsigned short* t;
    t = xhCur; xhCur = xhNxt; xhNxt = t;
    t = ehCur; ehCur = ehNxt; ehNxt = t;
  }

  final_kernel<<<wgsN, 128, 0, stream>>>(Nn, xhCur, (const uint4*)wFL1, fb1,
                                         (const uint4*)wFL2, fb2, (float*)d_out);
}
